// WindowedAtomBlock_27693949125122
// MI455X (gfx1250) — matrix operand order repaired and checked by reading, not compiled, not run on hardware
//
#include <hip/hip_runtime.h>

// ---------------- problem constants ----------------
#define B_    8
#define M_    8192
#define D_    128
#define NH_   4
#define DH_   32
#define NW_   256          // M_/32 windows
#define DFF_  512
#define R_TOT 65536        // B_*M_

// ---------------- vector types ----------------
typedef __bf16 v16bf __attribute__((ext_vector_type(16)));
typedef __bf16 v8bf  __attribute__((ext_vector_type(8)));
typedef __bf16 v4bf  __attribute__((ext_vector_type(4)));
typedef float  v8f   __attribute__((ext_vector_type(8)));
typedef unsigned int v4ui __attribute__((ext_vector_type(4)));
typedef int    v8si  __attribute__((ext_vector_type(8)));
typedef int    v4si  __attribute__((ext_vector_type(4)));

// ---------------- feature detection ----------------
#ifdef __HIP_DEVICE_COMPILE__
#if __has_builtin(__builtin_amdgcn_tensor_load_to_lds) && __has_builtin(__builtin_amdgcn_s_wait_tensorcnt)
#define HAVE_TDM 1
#endif
#endif
#ifndef HAVE_TDM
#define HAVE_TDM 0
#endif
#if defined(__clang_major__) && (__clang_major__ >= 23)
#define TDM_6ARG 1
#else
#define TDM_6ARG 0
#endif

// ---------------- small helpers ----------------
__device__ __forceinline__ float sigf(float x) { return 1.0f / (1.0f + __expf(-x)); }

__device__ __forceinline__ float wave_sum32(float x) {
  #pragma unroll
  for (int m = 1; m < 32; m <<= 1) x += __shfl_xor(x, m);
  return x;
}
__device__ __forceinline__ float rowsum16(float x) {
  x += __shfl_xor(x, 1); x += __shfl_xor(x, 2);
  x += __shfl_xor(x, 4); x += __shfl_xor(x, 8);
  return x;
}
__device__ __forceinline__ float rowmax16(float x) {
  x = fmaxf(x, __shfl_xor(x, 1)); x = fmaxf(x, __shfl_xor(x, 2));
  x = fmaxf(x, __shfl_xor(x, 4)); x = fmaxf(x, __shfl_xor(x, 8));
  return x;
}

// ---------------- WMMA wrappers ----------------
__device__ __forceinline__ v8f wmma_bf16(v16bf a, v16bf b, v8f c) {
  // D = A(16x32) * B(32x16) + C ; f32 accumulate
  return __builtin_amdgcn_wmma_f32_16x16x32_bf16(false, a, false, b, (short)0, c, false, false);
}

// A fragment: 16 rows x 32 K of bf16, row-major in LDS with row stride ld.
// Doc layout: lanes 0-15 = rows, K{0..7,16..23}; lanes 16-31 = rows, K{8..15,24..31}.
__device__ __forceinline__ v16bf load_a_frag(const __bf16* base, int ld) {
  const int lane = threadIdx.x & 31;
  const __bf16* p = base + (lane & 15) * ld + ((lane >> 4) << 3);
  v8bf lo = *(const v8bf*)p;
  v8bf hi = *(const v8bf*)(p + 16);
  v16bf r;
  #pragma unroll
  for (int i = 0; i < 8; ++i) { r[i] = lo[i]; r[8 + i] = hi[i]; }
  return r;
}

// B fragment: 32 K x 16 N, stored TRANSPOSED in LDS as [n][k] with row stride ld.
// lane = column N (mod 16); lanes 0-15 hold K{0..7,16..23}, lanes 16-31 hold K{8..15,24..31}.
__device__ __forceinline__ v16bf load_b_frag(const __bf16* baseT, int ld) {
  const int lane = threadIdx.x & 31;
  const __bf16* p = baseT + (lane & 15) * ld + ((lane >> 4) << 3);
  v8bf lo = *(const v8bf*)p;
  v8bf hi = *(const v8bf*)(p + 16);
  v16bf r;
  #pragma unroll
  for (int i = 0; i < 8; ++i) { r[i] = lo[i]; r[8 + i] = hi[i]; }
  return r;
}

// ---------------- TDM: 128x128 bf16 tile -> LDS (row pad 4 dwords / 64 dwords -> ld 136) ----------------
#if HAVE_TDM
__device__ __forceinline__ void tdm_load_tile_bf16(void* lds_dst, const void* gsrc, unsigned ld_elems) {
  v4ui g0 = {0u, 0u, 0u, 0u};
  v8si g1 = {0, 0, 0, 0, 0, 0, 0, 0};
  v4si gz = {0, 0, 0, 0};
  unsigned long long ga = (unsigned long long)gsrc;
  g0.x = 1u;                                            // count=1 (valid user descriptor)
  g0.y = (unsigned)(unsigned long long)lds_dst;         // lds byte address
  g0.z = (unsigned)ga;                                  // global addr [31:0]
  g0.w = (unsigned)((ga >> 32) & 0x01FFFFFFu) | (2u << 30); // addr [56:32] | type=2
  const unsigned pad_interval = 5u;                     // every 64 dwords (=128 bf16)
  const unsigned pad_amount   = 3u;                     // insert 4 dwords (=8 bf16) -> ld 136
  g1[0] = (int)((1u << 16) | (1u << 20) | (pad_interval << 22) | (pad_amount << 25)); // 2B elems, pad_enable
  g1[1] = (int)((ld_elems & 0xFFFFu) << 16);            // tensor_dim0 low16
  g1[2] = (int)(((ld_elems >> 16) & 0xFFFFu) | (8192u << 16)); // dim0 hi | tensor_dim1=8192
  g1[3] = (int)(128u << 16);                            // tile_dim0 = 128
  g1[4] = (int)(128u);                                  // tile_dim1 = 128
  g1[5] = (int)ld_elems;                                // tensor_dim0_stride low32
#if TDM_6ARG
  v8si z8 = {0, 0, 0, 0, 0, 0, 0, 0};
  __builtin_amdgcn_tensor_load_to_lds(g0, g1, gz, gz, z8, 0);
#else
  __builtin_amdgcn_tensor_load_to_lds(g0, g1, gz, gz, 0);
#endif
}
#endif

// ---------------- GEMM tile core (blockDim = 256 = 8 waves) ----------------
// Stage 128x128 bf16 activation panel, row-major, into LDS (ld 136)
__device__ __forceinline__ void stage_a_panel(const __bf16* Ag, long long row0, int lda, int k0, __bf16* sA) {
  __syncthreads();                    // guard against previous-iteration readers
  const int t = threadIdx.x;
  const int row = t >> 1;
  const int ch = (t & 1) << 6;        // 0 or 64
  const __bf16* src = Ag + (row0 + row) * (long long)lda + k0 + ch;
  __bf16* dst = sA + row * 136 + ch;
  #pragma unroll
  for (int i = 0; i < 8; ++i)
    *(v8bf*)(dst + i * 8) = *(const v8bf*)(src + i * 8);
}

// Stage 128(n) x 128(k) tile of a pre-transposed bf16 weight (WT is [N][K] row-major, row stride ldk)
__device__ __forceinline__ void stage_wt_panel(const __bf16* WTg, int n0, int k0, int ldk, __bf16* sWt) {
  __syncthreads();                    // all waves done reading previous sWt
#if HAVE_TDM
  if ((threadIdx.x >> 5) == 0) {
    tdm_load_tile_bf16(sWt, WTg + (long long)n0 * ldk + k0, (unsigned)ldk);
    __builtin_amdgcn_s_wait_tensorcnt(0);
  }
#else
  const int t = threadIdx.x;
  const int row = t >> 1;
  const int ch = (t & 1) << 6;
  const __bf16* src = WTg + (long long)(n0 + row) * ldk + k0 + ch;
  __bf16* dst = sWt + row * 136 + ch;
  #pragma unroll
  for (int i = 0; i < 8; ++i)
    *(v8bf*)(dst + i * 8) = *(const v8bf*)(src + i * 8);
#endif
  __syncthreads();                    // publish sA + sWt
}

// One wave: 16-row strip x 128 cols, K=128 panel -> 32 v_wmma
__device__ __forceinline__ void mma_panel(const __bf16* sA, const __bf16* sWt, v8f acc[8]) {
  const int wid = threadIdx.x >> 5;
  const __bf16* abase = sA + wid * 16 * 136;
  #pragma unroll
  for (int k0 = 0; k0 < 128; k0 += 32) {
    v16bf af = load_a_frag(abase + k0, 136);
    #pragma unroll
    for (int nt = 0; nt < 8; ++nt) {
      v16bf bf = load_b_frag(sWt + nt * 16 * 136 + k0, 136);
      acc[nt] = wmma_bf16(af, bf, acc[nt]);
    }
  }
}

// ==================== kernels ====================

// Transpose+convert W (KxN f32 row-major) -> WT (NxK bf16 row-major)
__global__ void k_wtrans(const float* __restrict__ W, __bf16* __restrict__ WT, int K, int N) {
  const int n = blockIdx.x;
  for (int k = threadIdx.x; k < K; k += blockDim.x)
    WT[(long long)n * K + k] = (__bf16)W[(long long)k * N + n];
}

// Row LN of a (f32 out) and of cond (bf16 out) + bf16 copy of cond. One wave per row.
__global__ void __launch_bounds__(256) k_prep(const float* __restrict__ a, const float* __restrict__ cond,
                                              float* __restrict__ ln_a, __bf16* __restrict__ sn_bf,
                                              __bf16* __restrict__ cond_bf) {
  const int wid = threadIdx.x >> 5, lane = threadIdx.x & 31;
  const long long r = (long long)blockIdx.x * 8 + wid;
  {
    float4 vv = ((const float4*)a)[r * 32 + lane];
    float mean = wave_sum32(vv.x + vv.y + vv.z + vv.w) * (1.f / 128.f);
    float dx = vv.x - mean, dy = vv.y - mean, dz = vv.z - mean, dw = vv.w - mean;
    float inv = rsqrtf(wave_sum32(dx * dx + dy * dy + dz * dz + dw * dw) * (1.f / 128.f) + 1e-5f);
    float4 o; o.x = dx * inv; o.y = dy * inv; o.z = dz * inv; o.w = dw * inv;
    ((float4*)ln_a)[r * 32 + lane] = o;
  }
  {
    float4 vv = ((const float4*)cond)[r * 32 + lane];
    float mean = wave_sum32(vv.x + vv.y + vv.z + vv.w) * (1.f / 128.f);
    float dx = vv.x - mean, dy = vv.y - mean, dz = vv.z - mean, dw = vv.w - mean;
    float inv = rsqrtf(wave_sum32(dx * dx + dy * dy + dz * dz + dw * dw) * (1.f / 128.f) + 1e-5f);
    v4bf sb = { (__bf16)(dx * inv), (__bf16)(dy * inv), (__bf16)(dz * inv), (__bf16)(dw * inv) };
    *(v4bf*)&sn_bf[(r * 32 + lane) * 4] = sb;
    v4bf cb = { (__bf16)vv.x, (__bf16)vv.y, (__bf16)vv.z, (__bf16)vv.w };
    *(v4bf*)&cond_bf[(r * 32 + lane) * 4] = cb;
  }
}

// Plain C = A(bf16 Rx128) @ W(128x128), f32 out
__global__ void __launch_bounds__(256) k_gemm_plain(const __bf16* __restrict__ A,
                                                    const __bf16* __restrict__ WT,
                                                    float* __restrict__ C) {
  __shared__ __bf16 sA[128 * 136];
  __shared__ __bf16 sWt[128 * 136];
  const long long row0 = (long long)blockIdx.x * 128;
  v8f acc[8] = {};
  stage_a_panel(A, row0, 128, 0, sA);
  stage_wt_panel(WT, 0, 0, 128, sWt);
  mma_panel(sA, sWt, acc);
  const int wid = threadIdx.x >> 5, lane = threadIdx.x & 31;
  const int rl = wid * 16 + ((lane >> 4) << 3);
  const int cl = lane & 15;
  #pragma unroll
  for (int nt = 0; nt < 8; ++nt)
    #pragma unroll
    for (int v = 0; v < 8; ++v)
      C[(row0 + rl + v) * 128 + nt * 16 + cl] = acc[nt][v];
}

// a_n = sigmoid(S1 + bs1)*ln_a + B1, bf16 out
__global__ void k_adaln(const float* __restrict__ S1, const float* __restrict__ B1,
                        const float* __restrict__ bs1, const float* __restrict__ ln_a,
                        __bf16* __restrict__ an) {
  const long long i = (long long)blockIdx.x * blockDim.x + threadIdx.x;  // over R*32 float4s
  float4 s = ((const float4*)S1)[i];
  float4 bm = ((const float4*)B1)[i];
  float4 l = ((const float4*)ln_a)[i];
  const int cb = ((int)(i & 31)) << 2;
  v4bf o = { (__bf16)(sigf(s.x + bs1[cb + 0]) * l.x + bm.x),
             (__bf16)(sigf(s.y + bs1[cb + 1]) * l.y + bm.y),
             (__bf16)(sigf(s.z + bs1[cb + 2]) * l.z + bm.z),
             (__bf16)(sigf(s.w + bs1[cb + 3]) * l.w + bm.w) };
  *(v4bf*)&an[i * 4] = o;
}

// Q/K/V/G projections; Q,K,V stored bf16 in [B][NH][M][DH]; G f32 [R][128]
__global__ void __launch_bounds__(256) k_qkvg(const __bf16* __restrict__ An,
      const __bf16* __restrict__ wqT, const __bf16* __restrict__ wkT,
      const __bf16* __restrict__ wvT, const __bf16* __restrict__ wgT,
      const float* __restrict__ bq,
      __bf16* __restrict__ Qb, __bf16* __restrict__ Kb, __bf16* __restrict__ Vb,
      float* __restrict__ G) {
  __shared__ __bf16 sA[128 * 136];
  __shared__ __bf16 sWt[128 * 136];
  const long long row0 = (long long)blockIdx.x * 128;
  stage_a_panel(An, row0, 128, 0, sA);
  const __bf16* Ws[4] = { wqT, wkT, wvT, wgT };
  __bf16* Os[3] = { Qb, Kb, Vb };
  const int wid = threadIdx.x >> 5, lane = threadIdx.x & 31;
  const int rl = wid * 16 + ((lane >> 4) << 3);
  const int cl = lane & 15;
  for (int s = 0; s < 4; ++s) {
    v8f acc[8] = {};
    stage_wt_panel(Ws[s], 0, 0, 128, sWt);
    mma_panel(sA, sWt, acc);
    #pragma unroll
    for (int nt = 0; nt < 8; ++nt) {
      #pragma unroll
      for (int v = 0; v < 8; ++v) {
        const long long r = row0 + rl + v;
        const int col = nt * 16 + cl;
        float x = acc[nt][v];
        if (s == 0) x += bq[col];
        if (s < 3) {
          const int h = col >> 5, dh = col & 31;
          const long long bb = r >> 13;          // / M_
          const long long mm = r & (M_ - 1);
          Os[s][(((bb * NH_ + h) * (long long)M_) + mm) * DH_ + dh] = (__bf16)x;
        } else {
          G[r * 128 + col] = x;
        }
      }
    }
  }
}

// Windowed attention: one 64-thread block per (b, h, window)
__global__ void __launch_bounds__(64) k_attn(const __bf16* __restrict__ Qb,
      const __bf16* __restrict__ Kb, const __bf16* __restrict__ Vb,
      const float* __restrict__ amask, __bf16* __restrict__ att) {
  __shared__ __bf16 qs[32 * 40];
  __shared__ __bf16 kbs[128 * 40];
  __shared__ __bf16 vts[32 * 136];
  __shared__ __bf16 ps[32 * 136];
  __shared__ float addm[128];
  const int t = threadIdx.x;
  const int wid = t >> 5, lane = t & 31;
  const int w = blockIdx.x;
  const int win = w & (NW_ - 1);
  const int h = (w >> 8) & (NH_ - 1);
  const int b = w >> 10;
  const long long head = (long long)(b * NH_ + h) * M_;
  { // stage Q (32x32)
    const int row = t >> 1, cc = (t & 1) << 4;
    const __bf16* src = Qb + (head + win * 32 + row) * DH_ + cc;
    *(v8bf*)&qs[row * 40 + cc] = *(const v8bf*)src;
    *(v8bf*)&qs[row * 40 + cc + 8] = *(const v8bf*)(src + 8);
  }
  // stage K rows, transposed V, additive mask
  for (int j = t; j < 128; j += 64) {
    const int srcm = win * 32 + j - 48;
    const bool ok = (srcm >= 0) && (srcm < M_);
    const int sc = ok ? srcm : 0;
    const float mk = ok ? amask[(long long)b * M_ + sc] : 0.f;
    addm[j] = (ok && mk > 0.f) ? 0.f : -1e9f;
    const __bf16* kr = Kb + (head + sc) * DH_;
    #pragma unroll
    for (int i = 0; i < 4; ++i)
      *(v8bf*)&kbs[j * 40 + i * 8] = *(const v8bf*)(kr + i * 8);
    const __bf16* vr = Vb + (head + sc) * DH_;
    #pragma unroll
    for (int d = 0; d < 32; ++d)
      vts[d * 136 + j] = vr[d];
  }
  __syncthreads();
  // scores: (16x32) @ (32x16) x 8 column tiles per wave
  const int r0 = wid * 16;
  const int mb = (lane >> 4) << 3;
  const int cl = lane & 15;
  v16bf aq = load_a_frag(qs + r0 * 40, 40);
  float p[8][8];
  const float scale = 0.17677669529663689f;  // 1/sqrt(32)
  #pragma unroll
  for (int nt = 0; nt < 8; ++nt) {
    v16bf bk = load_b_frag(kbs + nt * 16 * 40, 40);
    v8f z = {};
    v8f sc_ = wmma_bf16(aq, bk, z);
    const float am = addm[nt * 16 + cl];
    #pragma unroll
    for (int v = 0; v < 8; ++v) p[nt][v] = sc_[v] * scale + am;
  }
  // softmax along 128 keys (16 lanes x 8 tiles hold one row)
  #pragma unroll
  for (int v = 0; v < 8; ++v) {
    float m = -1e30f;
    #pragma unroll
    for (int nt = 0; nt < 8; ++nt) m = fmaxf(m, p[nt][v]);
    m = rowmax16(m);
    float s = 0.f;
    #pragma unroll
    for (int nt = 0; nt < 8; ++nt) { p[nt][v] = __expf(p[nt][v] - m); s += p[nt][v]; }
    s = rowsum16(s);
    const float inv = 1.f / s;
    #pragma unroll
    for (int nt = 0; nt < 8; ++nt)
      ps[(r0 + mb + v) * 136 + nt * 16 + cl] = (__bf16)(p[nt][v] * inv);
  }
  __syncthreads();
  // P @ V : K loop over 128 keys, 2 output tiles (DH=32)
  v8f o[2] = {};
  #pragma unroll
  for (int k0 = 0; k0 < 128; k0 += 32) {
    v16bf ap = load_a_frag(ps + r0 * 136 + k0, 136);
    #pragma unroll
    for (int nt = 0; nt < 2; ++nt) {
      v16bf bv = load_b_frag(vts + nt * 16 * 136 + k0, 136);
      o[nt] = wmma_bf16(ap, bv, o[nt]);
    }
  }
  #pragma unroll
  for (int nt = 0; nt < 2; ++nt)
    #pragma unroll
    for (int v = 0; v < 8; ++v) {
      const long long r = (long long)b * M_ + win * 32 + r0 + mb + v;
      att[r * 128 + h * 32 + nt * 16 + cl] = (__bf16)o[nt][v];
    }
}

// att @ wo + gated residual + row LN + adaLN2 combine (writes a1 f32 and x bf16)
__global__ void __launch_bounds__(256) k_attnout(const __bf16* __restrict__ att,
      const __bf16* __restrict__ woT,
      const float* __restrict__ G, const float* __restrict__ AG, const float* __restrict__ bag,
      const float* __restrict__ a_in, const float* __restrict__ amask,
      const float* __restrict__ S2, const float* __restrict__ B2, const float* __restrict__ bs2,
      float* __restrict__ a1, __bf16* __restrict__ xout) {
  __shared__ __bf16 sA[128 * 136];
  __shared__ __bf16 sWt[128 * 136];
  const long long row0 = (long long)blockIdx.x * 128;
  v8f acc[8] = {};
  stage_a_panel(att, row0, 128, 0, sA);
  stage_wt_panel(woT, 0, 0, 128, sWt);
  mma_panel(sA, sWt, acc);
  const int wid = threadIdx.x >> 5, lane = threadIdx.x & 31;
  const int rl = wid * 16 + ((lane >> 4) << 3);
  const int cl = lane & 15;
  float a1r[8][8];
  #pragma unroll
  for (int v = 0; v < 8; ++v) {
    const long long r = row0 + rl + v;
    const float mk = amask[r];
    #pragma unroll
    for (int nt = 0; nt < 8; ++nt) {
      const int col = nt * 16 + cl;
      const long long idx = r * 128 + col;
      const float au = sigf(G[idx]) * acc[nt][v];
      const float gate = sigf(AG[idx] + bag[col]);
      const float val = (a_in[idx] + gate * au) * mk;
      a1r[nt][v] = val;
      a1[idx] = val;
    }
  }
  #pragma unroll
  for (int v = 0; v < 8; ++v) {
    const long long r = row0 + rl + v;
    float s = 0.f;
    #pragma unroll
    for (int nt = 0; nt < 8; ++nt) s += a1r[nt][v];
    s = rowsum16(s);
    const float mean = s * (1.f / 128.f);
    float q = 0.f;
    #pragma unroll
    for (int nt = 0; nt < 8; ++nt) { const float d = a1r[nt][v] - mean; q += d * d; }
    q = rowsum16(q);
    const float inv = rsqrtf(q * (1.f / 128.f) + 1e-5f);
    #pragma unroll
    for (int nt = 0; nt < 8; ++nt) {
      const int col = nt * 16 + cl;
      const long long idx = r * 128 + col;
      const float ln = (a1r[nt][v] - mean) * inv;
      xout[idx] = (__bf16)(sigf(S2[idx] + bs2[col]) * ln + B2[idx]);
    }
  }
}

// H = silu(x@w1) * (x@w3), bf16 out (Rx512). blockIdx.y selects 128-col group.
__global__ void __launch_bounds__(256) k_ffn1(const __bf16* __restrict__ X,
      const __bf16* __restrict__ w1T, const __bf16* __restrict__ w3T,
      __bf16* __restrict__ H) {
  __shared__ __bf16 sA[128 * 136];
  __shared__ __bf16 sWt[128 * 136];
  const long long row0 = (long long)blockIdx.x * 128;
  const int c0 = blockIdx.y * 128;
  stage_a_panel(X, row0, 128, 0, sA);
  v8f acc1[8] = {};
  stage_wt_panel(w1T, c0, 0, 128, sWt);
  mma_panel(sA, sWt, acc1);
  v8f acc2[8] = {};
  stage_wt_panel(w3T, c0, 0, 128, sWt);
  mma_panel(sA, sWt, acc2);
  const int wid = threadIdx.x >> 5, lane = threadIdx.x & 31;
  const int rl = wid * 16 + ((lane >> 4) << 3);
  const int cl = lane & 15;
  #pragma unroll
  for (int nt = 0; nt < 8; ++nt)
    #pragma unroll
    for (int v = 0; v < 8; ++v) {
      const long long r = row0 + rl + v;
      const float u = acc1[nt][v];
      H[r * DFF_ + c0 + nt * 16 + cl] = (__bf16)((u * sigf(u)) * acc2[nt][v]);
    }
}

// out = (a1 + sigmoid(FG + bfg) * (H @ w2)) * mask ; K=512 -> 4 panels
__global__ void __launch_bounds__(256) k_ffn2(const __bf16* __restrict__ H,
      const __bf16* __restrict__ w2T,
      const float* __restrict__ FG, const float* __restrict__ bfg,
      const float* __restrict__ a1, const float* __restrict__ amask,
      float* __restrict__ out) {
  __shared__ __bf16 sA[128 * 136];
  __shared__ __bf16 sWt[128 * 136];
  const long long row0 = (long long)blockIdx.x * 128;
  v8f acc[8] = {};
  #pragma unroll
  for (int kp = 0; kp < 4; ++kp) {
    stage_a_panel(H, row0, DFF_, kp * 128, sA);
    if (kp < 3)
      __builtin_prefetch(H + (row0 + (threadIdx.x >> 1)) * DFF_ + (kp + 1) * 128, 0, 1);
    stage_wt_panel(w2T, 0, kp * 128, DFF_, sWt);
    mma_panel(sA, sWt, acc);
  }
  const int wid = threadIdx.x >> 5, lane = threadIdx.x & 31;
  const int rl = wid * 16 + ((lane >> 4) << 3);
  const int cl = lane & 15;
  #pragma unroll
  for (int v = 0; v < 8; ++v) {
    const long long r = row0 + rl + v;
    const float mk = amask[r];
    #pragma unroll
    for (int nt = 0; nt < 8; ++nt) {
      const int col = nt * 16 + cl;
      const long long idx = r * 128 + col;
      out[idx] = (a1[idx] + sigf(FG[idx] + bfg[col]) * acc[nt][v]) * mk;
    }
  }
}

// ==================== host launcher ====================
extern "C" void kernel_launch(void* const* d_in, const int* in_sizes, int n_in,
                              void* d_out, int out_size, void* d_ws, size_t ws_size,
                              hipStream_t stream) {
  (void)in_sizes; (void)n_in; (void)out_size; (void)ws_size;
  const float* a    = (const float*)d_in[0];
  const float* cond = (const float*)d_in[1];
  const float* amsk = (const float*)d_in[2];
  const float* ws1  = (const float*)d_in[3];
  const float* bs1  = (const float*)d_in[4];
  const float* wb1  = (const float*)d_in[5];
  const float* ws2  = (const float*)d_in[6];
  const float* bs2  = (const float*)d_in[7];
  const float* wb2  = (const float*)d_in[8];
  const float* wq   = (const float*)d_in[9];
  const float* bq   = (const float*)d_in[10];
  const float* wk   = (const float*)d_in[11];
  const float* wv   = (const float*)d_in[12];
  const float* wg   = (const float*)d_in[13];
  const float* wo   = (const float*)d_in[14];
  const float* wag  = (const float*)d_in[15];
  const float* bag  = (const float*)d_in[16];
  const float* wfg  = (const float*)d_in[17];
  const float* bfg  = (const float*)d_in[18];
  const float* w1   = (const float*)d_in[19];
  const float* w3   = (const float*)d_in[20];
  const float* w2   = (const float*)d_in[21];
  float* out = (float*)d_out;

  const size_t R = R_TOT;
  char* wsb = (char*)d_ws;
  size_t off = 0;
  auto take = [&](size_t bytes) -> char* {
    char* p = wsb + off;
    off = (off + bytes + 255) & ~(size_t)255;
    return p;
  };
  float*  ln_a    = (float*)take(R * 128 * 4);
  __bf16* sn_bf   = (__bf16*)take(R * 128 * 2);
  __bf16* cond_bf = (__bf16*)take(R * 128 * 2);
  float*  S1 = (float*)take(R * 128 * 4);
  float*  B1 = (float*)take(R * 128 * 4);
  float*  S2 = (float*)take(R * 128 * 4);
  float*  B2 = (float*)take(R * 128 * 4);
  float*  AG = (float*)take(R * 128 * 4);
  float*  FG = (float*)take(R * 128 * 4);
  __bf16* an_bf = (__bf16*)take(R * 128 * 2);
  __bf16* q_bf  = (__bf16*)take(R * 128 * 2);
  __bf16* k_bf  = (__bf16*)take(R * 128 * 2);
  __bf16* v_bf  = (__bf16*)take(R * 128 * 2);
  __bf16* h_bf  = (__bf16*)take(R * 512 * 2);
  // lifetime-disjoint aliases
  float*  Gbuf   = S1;     // G written after S1 consumed by k_adaln
  float*  a1buf  = B1;     // a1 written after B1 consumed by k_adaln
  __bf16* att_bf = an_bf;  // att written after an consumed by k_qkvg
  __bf16* x_bf   = sn_bf;  // x written after sn consumed by the 4 cond-GEMMs
  // transposed bf16 weights
  __bf16* ws1T = (__bf16*)take(128 * 128 * 2);
  __bf16* wb1T = (__bf16*)take(128 * 128 * 2);
  __bf16* ws2T = (__bf16*)take(128 * 128 * 2);
  __bf16* wb2T = (__bf16*)take(128 * 128 * 2);
  __bf16* wagT = (__bf16*)take(128 * 128 * 2);
  __bf16* wfgT = (__bf16*)take(128 * 128 * 2);
  __bf16* wqT  = (__bf16*)take(128 * 128 * 2);
  __bf16* wkT  = (__bf16*)take(128 * 128 * 2);
  __bf16* wvT  = (__bf16*)take(128 * 128 * 2);
  __bf16* wgT  = (__bf16*)take(128 * 128 * 2);
  __bf16* woT  = (__bf16*)take(128 * 128 * 2);
  __bf16* w1T  = (__bf16*)take(512 * 128 * 2);
  __bf16* w3T  = (__bf16*)take(512 * 128 * 2);
  __bf16* w2T  = (__bf16*)take(128 * 512 * 2);

  // 0) weight transpose/convert
  k_wtrans<<<128, 128, 0, stream>>>(ws1, ws1T, 128, 128);
  k_wtrans<<<128, 128, 0, stream>>>(wb1, wb1T, 128, 128);
  k_wtrans<<<128, 128, 0, stream>>>(ws2, ws2T, 128, 128);
  k_wtrans<<<128, 128, 0, stream>>>(wb2, wb2T, 128, 128);
  k_wtrans<<<128, 128, 0, stream>>>(wag, wagT, 128, 128);
  k_wtrans<<<128, 128, 0, stream>>>(wfg, wfgT, 128, 128);
  k_wtrans<<<128, 128, 0, stream>>>(wq,  wqT,  128, 128);
  k_wtrans<<<128, 128, 0, stream>>>(wk,  wkT,  128, 128);
  k_wtrans<<<128, 128, 0, stream>>>(wv,  wvT,  128, 128);
  k_wtrans<<<128, 128, 0, stream>>>(wg,  wgT,  128, 128);
  k_wtrans<<<128, 128, 0, stream>>>(wo,  woT,  128, 128);
  k_wtrans<<<512, 128, 0, stream>>>(w1,  w1T,  128, 512);
  k_wtrans<<<512, 128, 0, stream>>>(w3,  w3T,  128, 512);
  k_wtrans<<<128, 128, 0, stream>>>(w2,  w2T,  512, 128);

  // 1) layernorms + bf16 casts
  k_prep<<<R_TOT / 8, 256, 0, stream>>>(a, cond, ln_a, sn_bf, cond_bf);

  // 2) cond-derived GEMMs
  const int GB = R_TOT / 128;  // 512
  k_gemm_plain<<<GB, 256, 0, stream>>>(sn_bf,   ws1T, S1);
  k_gemm_plain<<<GB, 256, 0, stream>>>(sn_bf,   wb1T, B1);
  k_gemm_plain<<<GB, 256, 0, stream>>>(sn_bf,   ws2T, S2);
  k_gemm_plain<<<GB, 256, 0, stream>>>(sn_bf,   wb2T, B2);
  k_gemm_plain<<<GB, 256, 0, stream>>>(cond_bf, wagT, AG);
  k_gemm_plain<<<GB, 256, 0, stream>>>(cond_bf, wfgT, FG);

  // 3) adaLN-1 combine -> a_n (bf16)
  k_adaln<<<(R_TOT * 32) / 256, 256, 0, stream>>>(S1, B1, bs1, ln_a, an_bf);

  // 4) Q/K/V/G projections
  k_qkvg<<<GB, 256, 0, stream>>>(an_bf, wqT, wkT, wvT, wgT, bq, q_bf, k_bf, v_bf, Gbuf);

  // 5) windowed attention (8 b * 4 h * 256 windows)
  k_attn<<<B_ * NH_ * NW_, 64, 0, stream>>>(q_bf, k_bf, v_bf, amsk, att_bf);

  // 6) wo GEMM + gated residual + LN + adaLN-2 combine
  k_attnout<<<GB, 256, 0, stream>>>(att_bf, woT, Gbuf, AG, bag, a, amsk,
                                    S2, B2, bs2, a1buf, x_bf);

  // 7) SwiGLU FFN
  k_ffn1<<<dim3(GB, 4), 256, 0, stream>>>(x_bf, w1T, w3T, h_bf);
  k_ffn2<<<GB, 256, 0, stream>>>(h_bf, w2T, FG, bfg, a1buf, amsk, out);
}
